// graphNetwork_try_57389353009175
// MI455X (gfx1250) — matrix operand order repaired and checked by reading, not compiled, not run on hardware
//
#include <hip/hip_runtime.h>
#include <hip/hip_bf16.h>
#include <math.h>

// CDNA5 / gfx1250: wave32, WMMA 16x16x32 bf16 with f32 accumulate.
typedef __attribute__((ext_vector_type(16))) __bf16 v16bf;
typedef __attribute__((ext_vector_type(8)))  float  v8f;

#define LN_EPS 1e-5f

// All flat element indices in this problem fit in 32 bits (largest buffer is
// 128 x 800000 = 102.4M elements), so index math is done in unsigned 32-bit:
// the compiler can then use 32-bit muls + SGPR-base/VGPR-offset addressing
// instead of v_mul_u64 chains.

// ---------------------------------------------------------------------------
// Pre-swizzle a [O,C] f32 weight matrix into WMMA A-fragment order (bf16).
// Per ISA 7.12.2, 16-bit A 16x32 (MxK):
//   lane l: m = mt*16 + (l&15), half h = l>>4
//   element i (0..15): k = ks*32 + h*8 + (i<8 ? i : i+8)
// Fragments stored contiguously: [((ks*numMt + mt)*32 + lane)*16 + i]
// so the GEMM loads each A fragment as 32 contiguous bytes (2x b128).
// ---------------------------------------------------------------------------
__global__ void pack_wfrag(const float* __restrict__ W, __bf16* __restrict__ out,
                           int O, int C) {
  int numMt = O >> 4;
  int numKs = (C + 31) >> 5;
  int tot = numKs * numMt * 512;
  int t = blockIdx.x * blockDim.x + threadIdx.x;
  if (t >= tot) return;
  int i    = t & 15;
  int lane = (t >> 4) & 31;
  int mt   = (t >> 9) % numMt;
  int ks   = (t >> 9) / numMt;
  int h = lane >> 4;
  int m = mt * 16 + (lane & 15);
  int k = ks * 32 + h * 8 + ((i < 8) ? i : (i + 8));
  float v = (k < C) ? W[(unsigned)m * (unsigned)C + (unsigned)k] : 0.0f;
  out[t] = (__bf16)v;
}

// ---------------------------------------------------------------------------
// Fused WMMA GEMM: out[O,L] = W[O,C] x B[C,L] with fused B generation and
// fused epilogues.  REQUIRES L % 16 == 0 (true here: N=50000, E=800000).
// Block = 128 threads = 4 waves; each wave owns one 16-column strip and MTW
// M-tiles.  Tail waves exit uniformly before any memory op, so all remaining
// loads/stores are unconditional (no per-lane exec masking).
// ---------------------------------------------------------------------------
enum { BS_PLAIN = 0, BS_NORMTANH = 1, BS_EDGE = 2, BS_CONCAT3 = 3 };
enum { EP_STORE = 0, EP_STORE_STATS = 1, EP_AXPY = 2, EP_BIAS_ELU = 3, EP_BIAS_T = 4 };

template<int O, int C, int MTW, int BSRC, int EPI>
__global__ __launch_bounds__(128)
void gemm_wmma(const __bf16* __restrict__ Wf,
               const float* __restrict__ X,
               const float* __restrict__ P0,
               const float* __restrict__ P1,
               const float* __restrict__ P2,
               const int* __restrict__ gI,
               const int* __restrict__ gJ,
               const float* __restrict__ statsIn, float invCntIn,
               float* __restrict__ statsOut,
               const float* __restrict__ bias,
               float* __restrict__ out,
               float hcoef, int L, int NN)
{
  constexpr int numMtTot = O >> 4;
  constexpr int numKs    = (C + 31) >> 5;
  const int wave    = threadIdx.x >> 5;
  const int lane    = threadIdx.x & 31;
  const int colTile = blockIdx.x * 4 + wave;
  // Wave-uniform early exit: terminated waves satisfy s_barrier (ISA 6.2),
  // so the stats __syncthreads below remains safe.
  if (colTile >= (L >> 4)) return;
  const int mtBase = blockIdx.y * MTW;
  const unsigned uL   = (unsigned)L;
  const unsigned uNN  = (unsigned)NN;
  const unsigned ncol = (unsigned)colTile * 16u + (unsigned)(lane & 15);

  // Global layer-norm parameters (mean / rsqrt(var+eps)) from fused stats.
  float meanB = 0.f, invB = 0.f;
  if constexpr (BSRC == BS_NORMTANH) {
    float s = statsIn[0], ss = statsIn[1];
    meanB = s * invCntIn;
    float var = fmaxf(ss * invCntIn - meanB * meanB, 0.f);
    invB = rsqrtf(var + LN_EPS);
  }
  unsigned gi = 0, gj = 0;
  if constexpr (BSRC == BS_EDGE) { gi = (unsigned)gI[ncol]; gj = (unsigned)gJ[ncol]; }

  v8f acc[MTW] = {};

  for (int ks = 0; ks < numKs; ++ks) {
    // B fragment per ISA layout: lane l holds col n=l&15, k = (l>>4)*8 + (i<8 ? i : i+8).
    const unsigned kb = (unsigned)ks * 32u + (unsigned)(lane >> 4) * 8u;
    if constexpr (BSRC == BS_PLAIN || BSRC == BS_NORMTANH) {
      if (ks + 1 < numKs)
        __builtin_prefetch(&X[(kb + 32u) * uL + ncol], 0, 1);  // global_prefetch_b8
    }
    v16bf bfrag;
    #pragma unroll
    for (unsigned i = 0; i < 16; ++i) {
      const unsigned k = kb + ((i < 8u) ? i : (i + 8u));
      float v;
      if constexpr (BSRC == BS_PLAIN) {
        if constexpr (C % 32 != 0) {
          // clamp address, select-zero padding lanes (v_cndmask, no branch)
          unsigned kc = (k < (unsigned)C) ? k : (unsigned)(C - 1);
          float t = X[kc * uL + ncol];
          v = (k < (unsigned)C) ? t : 0.f;
        } else {
          v = X[k * uL + ncol];
        }
      } else if constexpr (BSRC == BS_NORMTANH) {
        v = tanhf((X[k * uL + ncol] - meanB) * invB);
      } else if constexpr (BSRC == BS_EDGE) {
        // Fused concat [aveX | xe | gradX] built from gathers; never
        // materialize the [192,E] tensor.
        if (k < 64u)       v = 0.5f * (P0[k * uNN + gi] + P0[k * uNN + gj]);
        else if (k < 128u) v = P1[(k - 64u) * uL + ncol];
        else               v = P0[(k - 128u) * uNN + gi] - P0[(k - 128u) * uNN + gj];
      } else { // BS_CONCAT3: [aveE | divE | xn]
        if (k < 64u)       v = P0[k * uL + ncol];
        else if (k < 128u) v = P1[(k - 64u) * uL + ncol];
        else               v = P2[(k - 128u) * uL + ncol];
      }
      bfrag[i] = (__bf16)v;
    }
    #pragma unroll
    for (int mt = 0; mt < MTW; ++mt) {
      const v16bf afrag =
          *(const v16bf*)(Wf + ((unsigned)(ks * numMtTot + mtBase + mt) * 32u +
                                (unsigned)lane) * 16u);
      acc[mt] = __builtin_amdgcn_wmma_f32_16x16x32_bf16(
          false, afrag, false, bfrag, (short)0, acc[mt], false, false);
    }
  }

  // Epilogue. C/D layout: lane l -> col n=l&15, rows m = (l>>4)*8 + r.
  float lsum = 0.f, lss = 0.f;
  const int mrb = (lane >> 4) * 8;
  #pragma unroll
  for (int mt = 0; mt < MTW; ++mt) {
    #pragma unroll
    for (int r = 0; r < 8; ++r) {
      const unsigned m = (unsigned)((mtBase + mt) * 16 + mrb + r);
      float v = acc[mt][r];
      if constexpr (EPI == EP_BIAS_ELU) { v += bias[m]; v = (v > 0.f) ? v : (expf(v) - 1.f); }
      if constexpr (EPI == EP_BIAS_T)   { v += bias[m]; }
      if constexpr (EPI == EP_AXPY)        out[m * uL + ncol] += hcoef * v;
      else if constexpr (EPI == EP_BIAS_T) out[ncol * (unsigned)O + m] = v;
      else                                 out[m * uL + ncol] = v;
      if constexpr (EPI == EP_STORE_STATS) { lsum += v; lss += v * v; }
    }
  }
  if constexpr (EPI == EP_STORE_STATS) {
    // Butterfly wave reduce (shfl_xor: no bounds checks -> lean lowering),
    // then 4 LDS atomics/block, then one global atomic pair per block.
    #pragma unroll
    for (int off = 16; off > 0; off >>= 1) {
      lsum += __shfl_xor(lsum, off, 32);
      lss  += __shfl_xor(lss,  off, 32);
    }
    __shared__ float sred[2];
    if (threadIdx.x == 0) { sred[0] = 0.f; sred[1] = 0.f; }
    __syncthreads();
    if (lane == 0) {
      atomicAdd(&sred[0], lsum);
      atomicAdd(&sred[1], lss);
    }
    __syncthreads();
    if (threadIdx.x == 0) {
      atomicAdd(&statsOut[0], sred[0]);
      atomicAdd(&statsOut[1], sred[1]);
    }
  }
}

// xe += h * (dxe - mean) * rsqrt(var+eps)   (post double-layer global LN)
__global__ void axpy_norm(float* __restrict__ y, const float* __restrict__ d,
                          const float* __restrict__ stats, float invCnt, float h, unsigned n) {
  unsigned i = blockIdx.x * blockDim.x + threadIdx.x;
  if (i >= n) return;
  float m   = stats[0] * invCnt;
  float inv = rsqrtf(fmaxf(stats[1] * invCnt - m * m, 0.f) + LN_EPS);
  y[i] += h * (d[i] - m) * inv;
}

// edge_div / edge_ave scatter: one thread per edge, loop over 64 channels.
__global__ void scatter_edges(const float* __restrict__ xe, const int* __restrict__ gI,
                              const int* __restrict__ gJ, float* __restrict__ divb,
                              float* __restrict__ aveb, int E, int NN) {
  unsigned e = blockIdx.x * blockDim.x + threadIdx.x;
  if (e >= (unsigned)E) return;
  unsigned i = (unsigned)gI[e], j = (unsigned)gJ[e];
  const unsigned uE = (unsigned)E, uNN = (unsigned)NN;
  #pragma unroll 4
  for (unsigned c = 0; c < 64u; ++c) {
    float g = xe[c * uE + e];
    atomicAdd(&divb[c * uNN + i],  g);
    atomicAdd(&divb[c * uNN + j], -g);
    atomicAdd(&aveb[c * uNN + i], 0.5f * g);
    atomicAdd(&aveb[c * uNN + j], 0.5f * g);
  }
}

// In-place row log-softmax, one block per row of y[N,K].
__global__ __launch_bounds__(256) void log_softmax_rows(float* __restrict__ y, int K) {
  const unsigned n = blockIdx.x;
  float* row = y + n * (unsigned)K;
  __shared__ float red[256];
  float mx = -INFINITY;
  for (int c = threadIdx.x; c < K; c += 256) mx = fmaxf(mx, row[c]);
  red[threadIdx.x] = mx; __syncthreads();
  for (int s = 128; s > 0; s >>= 1) {
    if (threadIdx.x < s) red[threadIdx.x] = fmaxf(red[threadIdx.x], red[threadIdx.x + s]);
    __syncthreads();
  }
  mx = red[0]; __syncthreads();
  float sum = 0.f;
  for (int c = threadIdx.x; c < K; c += 256) sum += expf(row[c] - mx);
  red[threadIdx.x] = sum; __syncthreads();
  for (int s = 128; s > 0; s >>= 1) {
    if (threadIdx.x < s) red[threadIdx.x] += red[threadIdx.x + s];
    __syncthreads();
  }
  float lse = mx + logf(red[0]);
  for (int c = threadIdx.x; c < K; c += 256) row[c] -= lse;
}

// ---------------------------------------------------------------------------

extern "C" void kernel_launch(void* const* d_in, const int* in_sizes, int n_in,
                              void* d_out, int out_size, void* d_ws, size_t ws_size,
                              hipStream_t stream) {
  (void)n_in; (void)out_size; (void)ws_size;
  const float* xn0 = (const float*)d_in[0];
  const float* xe0 = (const float*)d_in[1];
  const int*   I   = (const int*)d_in[2];
  const int*   J   = (const int*)d_in[3];
  const float* K1N = (const float*)d_in[4];
  const float* K2N = (const float*)d_in[5];
  const float* K1E = (const float*)d_in[6];
  const float* K2E = (const float*)d_in[7];
  const float* KNc = (const float*)d_in[8];
  const float* KE1 = (const float*)d_in[9];
  const float* KE2 = (const float*)d_in[10];
  const float* KN1 = (const float*)d_in[11];
  const float* KN2 = (const float*)d_in[12];
  const float* L1w = (const float*)d_in[13];
  const float* L1b = (const float*)d_in[14];
  const float* L2w = (const float*)d_in[15];
  const float* L2b = (const float*)d_in[16];

  const int N = in_sizes[0] / 16;
  const int E = in_sizes[1] / 16;
  const int nlayer = in_sizes[9] / (128 * 192);
  const float h = 0.1f;

  // ---- workspace carve-up (256B aligned) ----
  char* w = (char*)d_ws;
  auto alloc = [&](size_t bytes) -> char* {
    char* p = w; w += (bytes + 255) & ~(size_t)255; return p;
  };
  float*  xn    = (float*)alloc((size_t)64  * N * 4);
  float*  xe    = (float*)alloc((size_t)64  * E * 4);
  float*  hidE  = (float*)alloc((size_t)128 * E * 4);
  float*  dxe   = (float*)alloc((size_t)64  * E * 4);
  float*  hidN  = (float*)alloc((size_t)128 * N * 4);
  float*  aveb  = (float*)alloc((size_t)64  * N * 4);
  float*  divb  = (float*)alloc((size_t)64  * N * 4);
  float*  stats = (float*)alloc(64 * 2 * 4);
  __bf16* fK1N = (__bf16*)alloc(2048 * 2);
  __bf16* fK2N = (__bf16*)alloc(4096 * 2);
  __bf16* fK1E = (__bf16*)alloc(2048 * 2);
  __bf16* fK2E = (__bf16*)alloc(4096 * 2);
  __bf16* fKNc = (__bf16*)alloc(4096 * 2);
  __bf16* fKE1 = (__bf16*)alloc((size_t)nlayer * 24576 * 2);
  __bf16* fKE2 = (__bf16*)alloc((size_t)nlayer * 8192  * 2);
  __bf16* fKN1 = (__bf16*)alloc((size_t)nlayer * 24576 * 2);
  __bf16* fKN2 = (__bf16*)alloc((size_t)nlayer * 8192  * 2);
  __bf16* fL1  = (__bf16*)alloc(16384 * 2);
  __bf16* fL2  = (__bf16*)alloc(262144 * 2);
  // Final head buffers alias hidE (dead after the last layer).
  float* feat = hidE;
  float* z1   = hidE + (size_t)64 * N;

  auto pack = [&](const float* src, __bf16* dst, int O, int C) {
    int tot = ((C + 31) / 32) * (O / 16) * 512;
    pack_wfrag<<<(tot + 255) / 256, 256, 0, stream>>>(src, dst, O, C);
  };
  auto grd = [&](int L, int gy) { return dim3((unsigned)((L / 16 + 3) / 4), (unsigned)gy, 1); };

  hipMemsetAsync(stats, 0, 64 * 2 * 4, stream);

  pack(K1N, fK1N, 64, 16);   pack(K2N, fK2N, 64, 64);
  pack(K1E, fK1E, 64, 16);   pack(K2E, fK2E, 64, 64);
  pack(KNc, fKNc, 64, 64);
  for (int i = 0; i < nlayer; ++i) {
    pack(KE1 + (size_t)i * 128 * 192, fKE1 + (size_t)i * 24576, 128, 192);
    pack(KE2 + (size_t)i * 64  * 128, fKE2 + (size_t)i * 8192,  64,  128);
    pack(KN1 + (size_t)i * 128 * 192, fKN1 + (size_t)i * 24576, 128, 192);
    pack(KN2 + (size_t)i * 64  * 128, fKN2 + (size_t)i * 8192,  64,  128);
  }
  pack(L1w, fL1, 256, 64);
  pack(L2w, fL2, 1024, 256);

  // ---- opening double-layers ----
  gemm_wmma<64, 16, 4, BS_PLAIN, EP_STORE_STATS><<<grd(N, 1), 128, 0, stream>>>(
      fK1N, xn0, nullptr, nullptr, nullptr, nullptr, nullptr,
      nullptr, 0.f, stats + 0, nullptr, hidN, 0.f, N, N);
  gemm_wmma<64, 64, 4, BS_NORMTANH, EP_STORE><<<grd(N, 1), 128, 0, stream>>>(
      fK2N, hidN, nullptr, nullptr, nullptr, nullptr, nullptr,
      stats + 0, 1.f / (64.f * (float)N), nullptr, nullptr, xn, 0.f, N, N);

  gemm_wmma<64, 16, 4, BS_PLAIN, EP_STORE_STATS><<<grd(E, 1), 128, 0, stream>>>(
      fK1E, xe0, nullptr, nullptr, nullptr, nullptr, nullptr,
      nullptr, 0.f, stats + 2, nullptr, hidE, 0.f, E, N);
  gemm_wmma<64, 64, 4, BS_NORMTANH, EP_STORE><<<grd(E, 1), 128, 0, stream>>>(
      fK2E, hidE, nullptr, nullptr, nullptr, nullptr, nullptr,
      stats + 2, 1.f / (64.f * (float)E), nullptr, nullptr, xe, 0.f, E, N);

  // ---- graph layers ----
  for (int i = 0; i < nlayer; ++i) {
    float* sA = stats + 2 * (2 + 3 * i);
    float* sB = sA + 2;
    float* sC = sA + 4;

    // edge double-layer: GEMM1 fuses gather-concat; stats for global LN
    gemm_wmma<128, 192, 8, BS_EDGE, EP_STORE_STATS><<<grd(E, 1), 128, 0, stream>>>(
        fKE1 + (size_t)i * 24576, nullptr, xn, xe, nullptr, I, J,
        nullptr, 0.f, sA, nullptr, hidE, 0.f, E, N);
    // GEMM2 fuses LN+tanh on read; stats of dxe for the trailing LN
    gemm_wmma<64, 128, 4, BS_NORMTANH, EP_STORE_STATS><<<grd(E, 1), 128, 0, stream>>>(
        fKE2 + (size_t)i * 8192, hidE, nullptr, nullptr, nullptr, nullptr, nullptr,
        sA, 1.f / (128.f * (float)E), sB, nullptr, dxe, 0.f, E, N);
    // xe += h * LN(dxe)
    axpy_norm<<<(unsigned)(((size_t)64 * E + 255) / 256), 256, 0, stream>>>(
        xe, dxe, sB, 1.f / (64.f * (float)E), h, (unsigned)(64u * (unsigned)E));

    // edge_div / edge_ave scatters
    hipMemsetAsync(aveb, 0, (size_t)64 * N * 4, stream);
    hipMemsetAsync(divb, 0, (size_t)64 * N * 4, stream);
    scatter_edges<<<(E + 255) / 256, 256, 0, stream>>>(xe, I, J, divb, aveb, E, N);

    // node double-layer
    gemm_wmma<128, 192, 8, BS_CONCAT3, EP_STORE_STATS><<<grd(N, 1), 128, 0, stream>>>(
        fKN1 + (size_t)i * 24576, nullptr, aveb, divb, xn, nullptr, nullptr,
        nullptr, 0.f, sC, nullptr, hidN, 0.f, N, N);
    gemm_wmma<64, 128, 4, BS_NORMTANH, EP_AXPY><<<grd(N, 1), 128, 0, stream>>>(
        fKN2 + (size_t)i * 8192, hidN, nullptr, nullptr, nullptr, nullptr, nullptr,
        sC, 1.f / (128.f * (float)N), nullptr, nullptr, xn, h, N, N);
  }

  // ---- head: close conv, lin1 (bias+ELU), lin2 (bias, transposed store) ----
  gemm_wmma<64, 64, 4, BS_PLAIN, EP_STORE><<<grd(N, 1), 128, 0, stream>>>(
      fKNc, xn, nullptr, nullptr, nullptr, nullptr, nullptr,
      nullptr, 0.f, nullptr, nullptr, feat, 0.f, N, N);
  gemm_wmma<256, 64, 8, BS_PLAIN, EP_BIAS_ELU><<<grd(N, 2), 128, 0, stream>>>(
      fL1, feat, nullptr, nullptr, nullptr, nullptr, nullptr,
      nullptr, 0.f, nullptr, L1b, z1, 0.f, N, N);
  gemm_wmma<1024, 256, 8, BS_PLAIN, EP_BIAS_T><<<grd(N, 8), 128, 0, stream>>>(
      fL2, z1, nullptr, nullptr, nullptr, nullptr, nullptr,
      nullptr, 0.f, nullptr, L2b, (float*)d_out, 0.f, N, N);

  log_softmax_rows<<<N, 256, 0, stream>>>((float*)d_out, 1024);
}
